// WubuBlock_35905926595396
// MI455X (gfx1250) — matrix operand order repaired and checked by reading, not compiled, not run on hardware
//
#include <hip/hip_runtime.h>
#include <hip/hip_bf16.h>
#include <math.h>

typedef __attribute__((ext_vector_type(16))) _Float16 v16h;
typedef __attribute__((ext_vector_type(8)))  float    v8f;

#define EPSF   1e-7f
#define LNEPS  1e-6f

static constexpr int NSEQ  = 1024;
static constexpr int DIMC  = 1024;
static constexpr int NHEAD = 16;
static constexpr int DHEAD = 64;
static constexpr int FFD   = 4096;
static constexpr int BATCH = 2;
static constexpr int RROWS = BATCH * NSEQ;

// 16-byte plain aggregate (HIP's uint4 has ctors -> not union-safe)
struct __attribute__((aligned(16))) U4 { unsigned int x, y, z, w; };
union FragH { U4 u[2]; v16h h; };

// ---------------- reductions ----------------
__device__ __forceinline__ float block_sum_256(float v) {
  __shared__ float red[8];
  const int t = threadIdx.x;
#pragma unroll
  for (int o = 16; o > 0; o >>= 1) v += __shfl_xor(v, o, 32);
  __syncthreads();                       // protect red vs previous call
  if ((t & 31) == 0) red[t >> 5] = v;
  __syncthreads();
  float s = red[0];
#pragma unroll
  for (int i = 1; i < 8; ++i) s += red[i];
  return s;
}

__device__ __forceinline__ float block_sum_64(float v) {
  __shared__ float red[2];
  const int t = threadIdx.x;
#pragma unroll
  for (int o = 16; o > 0; o >>= 1) v += __shfl_xor(v, o, 32);
  __syncthreads();
  if ((t & 31) == 0) red[t >> 5] = v;
  __syncthreads();
  return red[0] + red[1];
}

__device__ __forceinline__ float gelu_tanh(float x) {
  return 0.5f * x * (1.0f + tanhf(0.7978845608028654f * (x + 0.044715f * x * x * x)));
}

// ---------------- weight convert: W[K][N] f32 -> Wt[N][K] f16 ----------------
__global__ void convert_wT_kernel(const float* __restrict__ W, _Float16* __restrict__ Wt,
                                  int K, int Nout) {
  int idx = blockIdx.x * 256 + threadIdx.x;
  if (idx >= K * Nout) return;
  int n = idx / K, k = idx - n * K;
  Wt[(size_t)n * K + k] = (_Float16)W[(size_t)k * Nout + n];
}

// ---------------- tangent-space LN: out = logmap0(expmap0(LN(logmap0(x)))) ----------------
__global__ void __launch_bounds__(256)
ln_tangent_kernel(const float* __restrict__ X, const float* __restrict__ cArr,
                  const float* __restrict__ scale, const float* __restrict__ bias,
                  _Float16* __restrict__ outH) {
  const int row = blockIdx.x;
  const int b   = row / NSEQ;
  const float c  = cArr[b];
  const float sc = fmaxf(sqrtf(c), EPSF);
  const float* x = X + (size_t)row * DIMC;
  const int t = threadIdx.x;

  float xv[4], s2 = 0.f;
#pragma unroll
  for (int i = 0; i < 4; ++i) { xv[i] = x[t + i * 256]; s2 += xv[i] * xv[i]; }
  s2 = block_sum_256(s2);

  // logmap0(x)
  const float xn   = sqrtf(s2);
  const float mag1 = atanhf(fminf(xn, 1.f - EPSF)) / sc;
  const float k1   = (xn < EPSF) ? 0.f : mag1 / fmaxf(xn, EPSF);

  float yv[4], ys = 0.f;
#pragma unroll
  for (int i = 0; i < 4; ++i) { yv[i] = xv[i] * k1; ys += yv[i]; }
  const float mu = block_sum_256(ys) * (1.f / DIMC);
  float vs = 0.f;
#pragma unroll
  for (int i = 0; i < 4; ++i) { float d = yv[i] - mu; vs += d * d; }
  const float var  = block_sum_256(vs) * (1.f / DIMC);
  const float rstd = rsqrtf(var + LNEPS);

  float tv[4], tn2 = 0.f;
#pragma unroll
  for (int i = 0; i < 4; ++i) {
    int idx = t + i * 256;
    tv[i] = (yv[i] - mu) * rstd * scale[idx] + bias[idx];
    tn2 += tv[i] * tv[i];
  }
  tn2 = block_sum_256(tn2);

  // expmap0 (+project) then logmap0, all via analytic norms
  const float vn   = sqrtf(tn2);
  const float safe = fmaxf(vn, EPSF);
  const float mag2 = tanhf(sc * safe) / sc;
  const float k2   = (vn < EPSF) ? 0.f : mag2 / safe;
  const float en   = k2 * vn;                                  // ||expmap0||
  const float pk   = (en >= 1.f) ? ((1.f - EPSF) / fmaxf(en, EPSF)) : 1.f;
  const float en2  = pk * en;                                  // post-projection norm
  const float mag3 = atanhf(fminf(en2, 1.f - EPSF)) / sc;
  const float k3   = (en2 < EPSF) ? 0.f : mag3 / fmaxf(en2, EPSF);
  const float ktot = k3 * pk * k2;

  _Float16* o = outH + (size_t)row * DIMC;
#pragma unroll
  for (int i = 0; i < 4; ++i) o[t + i * 256] = (_Float16)(ktot * tv[i]);
}

// ---------------- out = mobius_add(x, expmap0(tan), c) ----------------
__global__ void __launch_bounds__(256)
expmap_mobius_kernel(const float* __restrict__ Xin, const float* __restrict__ Tin,
                     const float* __restrict__ cArr, float* __restrict__ Out) {
  const int row = blockIdx.x;
  const int b   = row / NSEQ;
  const float c  = cArr[b];
  const float sc = fmaxf(sqrtf(c), EPSF);
  const float* xp = Xin + (size_t)row * DIMC;
  const float* op = Tin + (size_t)row * DIMC;
  const int t = threadIdx.x;

  float xv[4], ov[4];
  float so2 = 0.f, sx2 = 0.f, sxo = 0.f;
#pragma unroll
  for (int i = 0; i < 4; ++i) {
    xv[i] = xp[t + i * 256]; ov[i] = op[t + i * 256];
    so2 += ov[i] * ov[i]; sx2 += xv[i] * xv[i]; sxo += xv[i] * ov[i];
  }
  so2 = block_sum_256(so2);
  sx2 = block_sum_256(sx2);
  sxo = block_sum_256(sxo);

  // y = expmap0(o) (+project): y = ky * o
  const float on   = sqrtf(so2);
  const float safe = fmaxf(on, EPSF);
  const float mago = tanhf(sc * safe) / sc;
  const float ke   = (on < EPSF) ? 0.f : mago / safe;
  const float en   = ke * on;
  const float pk   = (en >= 1.f) ? ((1.f - EPSF) / fmaxf(en, EPSF)) : 1.f;
  const float ky   = pk * ke;
  const float y2   = (pk * en) * (pk * en);
  const float xy   = ky * sxo;

  // mobius_add(x, y)
  const float A   = 1.f + 2.f * c * xy + c * y2;
  const float Bc  = 1.f - c * sx2;
  const float den = fmaxf(1.f + 2.f * c * xy + c * c * sx2 * y2, EPSF);
  const float nn  = sqrtf(fmaxf(A * A * sx2 + 2.f * A * Bc * xy + Bc * Bc * y2, 0.f));
  const float zn  = nn / den;
  const float pz  = (zn >= 1.f) ? ((1.f - EPSF) / fmaxf(zn, EPSF)) : 1.f;
  const float sA  = pz * A / den, sB = pz * Bc * ky / den;

  float* out = Out + (size_t)row * DIMC;
#pragma unroll
  for (int i = 0; i < 4; ++i) out[t + i * 256] = sA * xv[i] + sB * ov[i];
}

// ---------------- RoPE + expmap0, in place, one block per (b,n,h) head ----------------
__global__ void __launch_bounds__(64)
rope_expmap_kernel(float* __restrict__ Q, const float* __restrict__ freqs,
                   const float* __restrict__ cArr) {
  const int blk = blockIdx.x;
  const int h = blk % NHEAD;
  const int n = (blk / NHEAD) % NSEQ;
  const int b = blk / (NHEAD * NSEQ);
  const float c  = cArr[b];
  const float sc = fmaxf(sqrtf(c), EPSF);
  float* p = Q + ((size_t)(b * NSEQ + n)) * DIMC + h * DHEAD;

  const int d = threadIdx.x;
  const int f = d & 31;
  const float tr = p[f], ti = p[32 + f];
  const float ang = freqs[n * 32 + f];
  const float cs = cosf(ang), sn = sinf(ang);
  const float val = (d < 32) ? (tr * cs - ti * sn) : (tr * sn + ti * cs);
  __syncthreads();                       // all reads before in-place writes

  const float n2   = block_sum_64(val * val);
  const float vn   = sqrtf(n2);
  const float safe = fmaxf(vn, EPSF);
  const float mag  = tanhf(sc * safe) / sc;
  const float k    = (vn < EPSF) ? 0.f : mag / safe;
  const float en   = k * vn;
  const float pk   = (en >= 1.f) ? ((1.f - EPSF) / fmaxf(en, EPSF)) : 1.f;
  p[d] = pk * k * val;
}

// ---------------- sliding-window Poincare attention: 1 wave per query ----------------
__global__ void __launch_bounds__(256)
attn_kernel(const float* __restrict__ QH, const float* __restrict__ KH,
            const float* __restrict__ V, const float* __restrict__ cArr,
            const float* __restrict__ geo, _Float16* __restrict__ OutH) {
  const int wid  = threadIdx.x >> 5;
  const int lane = threadIdx.x & 31;
  const int q    = blockIdx.x * 8 + wid;          // (b*H + h)*N + n
  const int n  = q % NSEQ;
  const int hh = (q / NSEQ) % NHEAD;
  const int b  = q / (NSEQ * NHEAD);
  const float c  = cArr[b];
  const float sc = fmaxf(sqrtf(c), EPSF);

  const size_t qoff = ((size_t)(b * NSEQ + n)) * DIMC + hh * DHEAD;
  const int  kidx  = n - 31 + lane;               // this lane's window key
  const bool valid = (kidx >= 0);
  const size_t koff = valid ? ((size_t)(b * NSEQ + kidx)) * DIMC + hh * DHEAD : 0;

  float x2 = 0.f, y2 = 0.f, xy = 0.f;
#pragma unroll 8
  for (int d = 0; d < DHEAD; ++d) {
    const float qd = QH[qoff + d];                // broadcast across lanes
    const float yd = valid ? KH[koff + d] : 0.f;  // padded key = 0 vector
    x2 += qd * qd; y2 += yd * yd; xy += qd * yd;
  }
  // ||mobius_add(-q, k)|| analytically: num = A*(-q) + B*k
  const float A   = 1.f - 2.f * c * xy + c * y2;
  const float Bc  = 1.f - c * x2;
  const float den = fmaxf(1.f - 2.f * c * xy + c * c * x2 * y2, EPSF);
  const float nn2 = fmaxf(A * A * x2 - 2.f * A * Bc * xy + Bc * Bc * y2, 0.f);
  float zn = sqrtf(nn2) / den;
  zn = (zn >= 1.f) ? (1.f - EPSF) : zn;           // project()
  const float arg  = fminf(sc * zn, 1.f - EPSF);
  const float dist = 2.f * atanhf(arg) / sc;
  const float score = -geo[hh] * dist;

  // wave softmax over the 32 window slots
  float m = score;
#pragma unroll
  for (int o = 16; o > 0; o >>= 1) m = fmaxf(m, __shfl_xor(m, o, 32));
  const float p = __expf(score - m);
  float s = p;
#pragma unroll
  for (int o = 16; o > 0; o >>= 1) s += __shfl_xor(s, o, 32);
  const float w = p / s;

  // weighted V: lane owns dims {lane, lane+32}
  float o0 = 0.f, o1 = 0.f;
#pragma unroll 4
  for (int t = 0; t < 32; ++t) {
    const float wt = __shfl(w, t, 32);
    const int kk = n - 31 + t;
    if (kk >= 0) {
      const size_t voff = ((size_t)(b * NSEQ + kk)) * DIMC + hh * DHEAD;
      o0 += wt * V[voff + lane];
      o1 += wt * V[voff + 32 + lane];
    }
  }
  const size_t obase = ((size_t)(b * NSEQ + n)) * DIMC + hh * DHEAD;
  OutH[obase + lane]      = (_Float16)o0;
  OutH[obase + 32 + lane] = (_Float16)o1;
}

// ---------------- WMMA GEMM: C[M][N] = act(A[M][K] @ Bt[N][K]^T + bias) ----------------
// Double-buffered LDS; tiles fetched with gfx1250 async global->LDS copies
// (GLOBAL_LOAD_ASYNC_TO_LDS_B128, ASYNCcnt-tracked), so no VGPR staging and the
// next K tile is in flight while the current one feeds the WMMAs.
template <int ACT, typename OutT>
__global__ void __launch_bounds__(256)
gemm_wmma(const _Float16* __restrict__ A, const _Float16* __restrict__ Bt,
          const float* __restrict__ bias, OutT* __restrict__ C,
          int M, int K, int Nout) {
  __shared__ _Float16 As[2][128][40];   // 32 K + 8 pad halves (80B stride, 16B aligned)
  __shared__ _Float16 Bs[2][128][40];

  const int tid  = threadIdx.x;
  const int lane = tid & 31;
  const int wave = tid >> 5;
  const int wm   = wave & 3;         // M sub-tile (32 rows)
  const int wn   = wave >> 2;        // N sub-tile (64 cols)
  const int m0   = blockIdx.x * 128;
  const int n0   = blockIdx.y * 128;
  const int l15  = lane & 15;
  const int kh   = lane >> 4;

  v8f acc[2][4];
#pragma unroll
  for (int i = 0; i < 2; ++i)
#pragma unroll
    for (int j = 0; j < 4; ++j) acc[i][j] = (v8f){0.f, 0.f, 0.f, 0.f, 0.f, 0.f, 0.f, 0.f};

  const int lrow  = tid >> 1;        // 0..127: row of the tile this thread fetches
  const int lhalf = (tid & 1) * 16;  // halves 0..15 or 16..31

  const _Float16* gA0 = A  + (size_t)(m0 + lrow) * K + lhalf;
  const _Float16* gB0 = Bt + (size_t)(n0 + lrow) * K + lhalf;

  // Issue async global->LDS copies for one 128x32 A tile + B tile into buffer `buf`.
  auto issue_tile = [&](int k0, int buf) {
    const _Float16* ga = gA0 + k0;
    const _Float16* gb = gB0 + k0;
    unsigned la0 = (unsigned)(uintptr_t)&As[buf][lrow][lhalf];
    unsigned la1 = (unsigned)(uintptr_t)&As[buf][lrow][lhalf + 8];
    unsigned lb0 = (unsigned)(uintptr_t)&Bs[buf][lrow][lhalf];
    unsigned lb1 = (unsigned)(uintptr_t)&Bs[buf][lrow][lhalf + 8];
    asm volatile("global_load_async_to_lds_b128 %0, %1, off" :: "v"(la0), "v"(ga)     : "memory");
    asm volatile("global_load_async_to_lds_b128 %0, %1, off" :: "v"(la1), "v"(ga + 8) : "memory");
    asm volatile("global_load_async_to_lds_b128 %0, %1, off" :: "v"(lb0), "v"(gb)     : "memory");
    asm volatile("global_load_async_to_lds_b128 %0, %1, off" :: "v"(lb1), "v"(gb + 8) : "memory");
  };

  const int nIter = K >> 5;

  // prologue: first tile
  issue_tile(0, 0);
  asm volatile("s_wait_asynccnt 0" ::: "memory");
  __syncthreads();

  for (int it = 0; it < nIter; ++it) {
    const int buf = it & 1;
    if (it + 1 < nIter) issue_tile((it + 1) * 32, buf ^ 1);  // overlap with compute
    if (it + 2 < nIter) {                                    // warm L2 one more tile out
      __builtin_prefetch((const void*)(gA0 + (it + 2) * 32), 0, 1);
      __builtin_prefetch((const void*)(gB0 + (it + 2) * 32), 0, 1);
    }

    // A fragments: lane l -> row l&15; lanes 0-15 hold K {0-7,16-23}, lanes 16-31 {8-15,24-31}
    FragH af[2];
#pragma unroll
    for (int tm = 0; tm < 2; ++tm) {
      const _Float16* ap = &As[buf][wm * 32 + tm * 16 + l15][0];
      af[tm].u[0] = *(const U4*)(ap + kh * 8);
      af[tm].u[1] = *(const U4*)(ap + 16 + kh * 8);
    }
#pragma unroll
    for (int tn = 0; tn < 4; ++tn) {
      // B fragment: lane l -> col l&15; lanes 0-15 K {0-7,16-23}, lanes 16-31 {8-15,24-31}
      FragH bf;
      const _Float16* bp = &Bs[buf][wn * 64 + tn * 16 + l15][0];
      bf.u[0] = *(const U4*)(bp + kh * 8);
      bf.u[1] = *(const U4*)(bp + 16 + kh * 8);
#pragma unroll
      for (int tm = 0; tm < 2; ++tm) {
        acc[tm][tn] = __builtin_amdgcn_wmma_f32_16x16x32_f16(
            false, af[tm].h, false, bf.h, (short)0, acc[tm][tn], false, false);
      }
    }

    if (it + 1 < nIter) {
      asm volatile("s_wait_asynccnt 0" ::: "memory");  // next tile's LDS writes landed
      __syncthreads();                                 // visible to all waves
    }
  }

  // epilogue: lane l -> rows (l>>4)*8 + i, col l&15 within 16x16 tile
#pragma unroll
  for (int tm = 0; tm < 2; ++tm)
#pragma unroll
    for (int tn = 0; tn < 4; ++tn) {
      const int col = n0 + wn * 64 + tn * 16 + l15;
      const float bb = bias[col];
#pragma unroll
      for (int i = 0; i < 8; ++i) {
        const int row = m0 + wm * 32 + tm * 16 + kh * 8 + i;
        float v = acc[tm][tn][i] + bb;
        if (ACT == 1) v = gelu_tanh(v);
        C[(size_t)row * Nout + col] = (OutT)v;
      }
    }
}

// ---------------- launch ----------------
extern "C" void kernel_launch(void* const* d_in, const int* in_sizes, int n_in,
                              void* d_out, int out_size, void* d_ws, size_t ws_size,
                              hipStream_t stream) {
  (void)in_sizes; (void)n_in; (void)out_size; (void)ws_size;
  const float* x     = (const float*)d_in[0];
  const float* freqs = (const float*)d_in[1];
  const float* c     = (const float*)d_in[2];
  const float* Wq = (const float*)d_in[3];   const float* bq = (const float*)d_in[4];
  const float* Wk = (const float*)d_in[5];   const float* bk = (const float*)d_in[6];
  const float* Wv = (const float*)d_in[7];   const float* bv = (const float*)d_in[8];
  const float* Wo = (const float*)d_in[9];   const float* bo = (const float*)d_in[10];
  const float* W1 = (const float*)d_in[11];  const float* b1 = (const float*)d_in[12];
  const float* W2 = (const float*)d_in[13];  const float* b2 = (const float*)d_in[14];
  const float* ln1s = (const float*)d_in[15]; const float* ln1b = (const float*)d_in[16];
  const float* ln2s = (const float*)d_in[17]; const float* ln2b = (const float*)d_in[18];
  const float* geo  = (const float*)d_in[19];

  char* ws = (char*)d_ws;
  constexpr size_t MB = 1ull << 20;
  _Float16* xtan_h = (_Float16*)(ws + 0);        // 4MB (reused as t2_h)
  _Float16* wq_h   = (_Float16*)(ws + 4 * MB);   // 2MB each
  _Float16* wk_h   = (_Float16*)(ws + 6 * MB);
  _Float16* wv_h   = (_Float16*)(ws + 8 * MB);
  _Float16* wo_h   = (_Float16*)(ws + 10 * MB);
  _Float16* w1_h   = (_Float16*)(ws + 12 * MB);  // 8MB
  _Float16* w2_h   = (_Float16*)(ws + 20 * MB);  // 8MB
  float*    qf     = (float*)(ws + 28 * MB);     // 8MB (reused: o_f32, ffn_f32)
  float*    kf     = (float*)(ws + 36 * MB);     // 8MB (reused: h1 lo)
  float*    vf     = (float*)(ws + 44 * MB);     // 8MB (reused: h1 hi)
  _Float16* attn_h = (_Float16*)(ws + 52 * MB);  // 4MB
  float*    xatt   = (float*)(ws + 56 * MB);     // 8MB
  float*    of32   = qf;
  _Float16* h1     = (_Float16*)(ws + 36 * MB);  // 16MB (k+v region)
  float*    ffn    = qf;
  _Float16* t2_h   = xtan_h;

  // 1) weight conversion (f32 -> transposed f16)
  convert_wT_kernel<<<(DIMC * DIMC) / 256, 256, 0, stream>>>(Wq, wq_h, DIMC, DIMC);
  convert_wT_kernel<<<(DIMC * DIMC) / 256, 256, 0, stream>>>(Wk, wk_h, DIMC, DIMC);
  convert_wT_kernel<<<(DIMC * DIMC) / 256, 256, 0, stream>>>(Wv, wv_h, DIMC, DIMC);
  convert_wT_kernel<<<(DIMC * DIMC) / 256, 256, 0, stream>>>(Wo, wo_h, DIMC, DIMC);
  convert_wT_kernel<<<(DIMC * FFD) / 256, 256, 0, stream>>>(W1, w1_h, DIMC, FFD);
  convert_wT_kernel<<<(FFD * DIMC) / 256, 256, 0, stream>>>(W2, w2_h, FFD, DIMC);

  // 2) tangent-space LN1
  ln_tangent_kernel<<<RROWS, 256, 0, stream>>>(x, c, ln1s, ln1b, xtan_h);

  // 3) QKV projections (WMMA)
  dim3 gP(RROWS / 128, DIMC / 128);
  gemm_wmma<0, float><<<gP, 256, 0, stream>>>(xtan_h, wq_h, bq, qf, RROWS, DIMC, DIMC);
  gemm_wmma<0, float><<<gP, 256, 0, stream>>>(xtan_h, wk_h, bk, kf, RROWS, DIMC, DIMC);
  gemm_wmma<0, float><<<gP, 256, 0, stream>>>(xtan_h, wv_h, bv, vf, RROWS, DIMC, DIMC);

  // 4) RoPE + expmap0 (in place)
  rope_expmap_kernel<<<BATCH * NSEQ * NHEAD, 64, 0, stream>>>(qf, freqs, c);
  rope_expmap_kernel<<<BATCH * NSEQ * NHEAD, 64, 0, stream>>>(kf, freqs, c);

  // 5) sliding-window hyperbolic attention
  attn_kernel<<<(BATCH * NHEAD * NSEQ) / 8, 256, 0, stream>>>(qf, kf, vf, c, geo, attn_h);

  // 6) output projection + Mobius residual + LN2
  gemm_wmma<0, float><<<gP, 256, 0, stream>>>(attn_h, wo_h, bo, of32, RROWS, DIMC, DIMC);
  expmap_mobius_kernel<<<RROWS, 256, 0, stream>>>(x, of32, c, xatt);
  ln_tangent_kernel<<<RROWS, 256, 0, stream>>>(xatt, c, ln2s, ln2b, t2_h);

  // 7) FFN (WMMA, fused GELU) + final Mobius residual
  dim3 gF1(RROWS / 128, FFD / 128);
  gemm_wmma<1, _Float16><<<gF1, 256, 0, stream>>>(t2_h, w1_h, b1, h1, RROWS, DIMC, FFD);
  dim3 gF2(RROWS / 128, DIMC / 128);
  gemm_wmma<0, float><<<gF2, 256, 0, stream>>>(h1, w2_h, b2, ffn, RROWS, FFD, DIMC);
  expmap_mobius_kernel<<<RROWS, 256, 0, stream>>>(xatt, ffn, c, (float*)d_out);
}
